// KANTLinear_45414984188167
// MI455X (gfx1250) — compile-verified, matrix operand order checked by reading, not run on hardware
//
#include <hip/hip_runtime.h>

// ---------------------------------------------------------------------------
// KAN (Chebyshev) layer, fused as one virtual GEMM on CDNA5 WMMA (bf16 -> f32)
//   out[r,n] = silu(x[r,:]) . Wb[n,:]  +  0.1 * Fl[r,:] . Wc_flat[n,:]
//   Fl[r, g*512+i] = T_o(x[b,i]),  m = 10r+g, b = m & 16383, o = m >> 14
// M=16384, N=512, K = 512 + 5120 = 5632
//
// Pipeline (TDM path):
//   prologue kernel: Wb|0.1*Wc -> bf16 workspace (once, not 128x per WG)
//   main kernel:     B tile   : tensor_load_to_lds (TDM DMA, padded stride)
//                    A tile   : x loads early, SiLU/Chebyshev store late
//                               (batched recurrence: one loop per thread)
//                    compute  : 16x v_wmma_f32_16x16x32_bf16 per K-chunk
//                    1 barrier / chunk, A+B double buffered via LDS offsets
// ---------------------------------------------------------------------------

typedef __attribute__((ext_vector_type(16))) __bf16         v16bf;
typedef __attribute__((ext_vector_type(8)))  float          v8f;
typedef __attribute__((ext_vector_type(4)))  unsigned int   u32x4;
typedef __attribute__((ext_vector_type(8)))  int            i32x8;
typedef __attribute__((ext_vector_type(4)))  int            i32x4;
typedef __attribute__((ext_vector_type(4)))  float          f32x4;
typedef __attribute__((ext_vector_type(8)))  unsigned short u16x8;
typedef __attribute__((ext_vector_type(4)))  unsigned short u16x4;

#define B_ROWS   16384
#define IN_DIM   512
#define OUT_DIM  512
#define ORDER    10
#define K_TOTAL  (IN_DIM + IN_DIM * ORDER)   // 5632
#define NCHUNK   (K_TOTAL / 32)              // 176
#define M_BLK    128
#define N_BLK    512
#define K_CHUNK  32                          // == WMMA K
#define LDS_LD   40                          // padded stride in halves (mult of 8)
#define THREADS  512                         // 16 waves (wave32)

#define BSZ      (N_BLK * LDS_LD)            // halves per B buffer
#define ASZ      (M_BLK * LDS_LD)            // halves per A buffer

#define WS_BYTES ((size_t)OUT_DIM * K_TOTAL * 2)  // 5,767,168 B bf16 weights

struct FragBits { u32x4 lo, hi; };           // 32 bytes == v16bf

__device__ __forceinline__ unsigned short f2bf(float f) {
  unsigned int u = __float_as_uint(f);
  u += 0x7FFFu + ((u >> 16) & 1u);           // round-to-nearest-even
  return (unsigned short)(u >> 16);
}

__device__ __forceinline__ float silu_f(float v) {
  return v / (1.0f + __expf(-v));
}

__device__ __forceinline__ v16bf load_frag16(const unsigned short* p) {
  FragBits fb;
  fb.lo = *(const u32x4*)(p);        // K 0-7  (lanes 0-15) / K 8-15  (16-31)
  fb.hi = *(const u32x4*)(p + 16);   // K 16-23 (lanes 0-15) / K 24-31 (16-31)
  return __builtin_bit_cast(v16bf, fb);
}

// ---------------------------------------------------------------------------
// A-tile staging, split so global loads can issue before the WMMAs
// ---------------------------------------------------------------------------
struct ASrc {
  f32x4 v0, v1;
  int   o;       // Chebyshev order (uniform across this thread's 8 elems)
  bool  base;
};

__device__ __forceinline__ ASrc load_a_src(int kc, const float* __restrict__ x,
                                           int r0, int aRowF, int aColF) {
  ASrc s;
  const int r = r0 + aRowF;
  const float* src;
  s.base = (kc < IN_DIM);
  s.o = 0;
  if (s.base) {
    src = x + (size_t)r * IN_DIM + kc + aColF;
  } else {
    const int c     = kc - IN_DIM;
    const int g     = c >> 9;              // chunk within one g segment
    const int iBase = c & (IN_DIM - 1);
    const int m     = r * ORDER + g;
    const int b     = m & (B_ROWS - 1);
    s.o = m >> 14;
    src = x + (size_t)b * IN_DIM + iBase + aColF;
  }
  s.v0 = *(const f32x4*)(src);
  s.v1 = *(const f32x4*)(src + 4);
  return s;
}

__device__ __forceinline__ void store_a_tile(unsigned short* dst, ASrc s) {
  float v[8];
#pragma unroll
  for (int j = 0; j < 4; ++j) {
    v[j]     = s.v0[j];
    v[4 + j] = s.v1[j];
  }
  float r[8];
  if (s.base) {
#pragma unroll
    for (int j = 0; j < 8; ++j) r[j] = silu_f(v[j]);
  } else {
    // T_o for 8 elems in lock-step: o is thread-uniform, single loop,
    // 8 independent FMAs per trip (dual-issue friendly, one branch).
    float t0[8], t1[8];
#pragma unroll
    for (int j = 0; j < 8; ++j) { t0[j] = 1.0f; t1[j] = v[j]; }
    for (int itr = 1; itr < s.o; ++itr) {
#pragma unroll
      for (int j = 0; j < 8; ++j) {
        const float t2 = __builtin_fmaf(2.0f * v[j], t1[j], -t0[j]);
        t0[j] = t1[j];
        t1[j] = t2;
      }
    }
#pragma unroll
    for (int j = 0; j < 8; ++j) r[j] = (s.o == 0) ? 1.0f : t1[j];
  }
  u16x8 t;
#pragma unroll
  for (int j = 0; j < 8; ++j) t[j] = f2bf(r[j]);
  *(u16x8*)dst = t;                          // one 16B ds_store
}

// ---------------------------------------------------------------------------
// Prologue: convert (Wb | 0.1*Wc) to bf16 row-major [OUT_DIM][K_TOTAL]
// ---------------------------------------------------------------------------
__global__ __launch_bounds__(256)
void convert_weights(const float* __restrict__ wb,
                     const float* __restrict__ wc,
                     unsigned short* __restrict__ dst) {
  const int idx = blockIdx.x * blockDim.x + threadIdx.x;   // one per 4 elems
  const int total = OUT_DIM * K_TOTAL / 4;
  if (idx >= total) return;
  const int n = (idx * 4) / K_TOTAL;
  const int k = (idx * 4) % K_TOTAL;
  f32x4 v;
  float scale;
  if (k < IN_DIM) {
    v = *(const f32x4*)(wb + (size_t)n * IN_DIM + k);
    scale = 1.0f;
  } else {
    v = *(const f32x4*)(wc + (size_t)n * (IN_DIM * ORDER) + (k - IN_DIM));
    scale = 0.1f;                    // the reference's /ORDER
  }
  u16x4 o;
  o.x = f2bf(v.x * scale);
  o.y = f2bf(v.y * scale);
  o.z = f2bf(v.z * scale);
  o.w = f2bf(v.w * scale);
  *(u16x4*)(dst + (size_t)idx * 4) = o;
}

// ---------------------------------------------------------------------------
// TDM: load one 32(K) x 512(N) bf16 weight tile into LDS with stride 40 halves
//   tensor: [OUT_DIM rows][K_TOTAL cols] bf16, row stride K_TOTAL elems
//   tile:   tile_dim0 = 32 elems (contig K), tile_dim1 = 512 rows (N)
//   pad:    every 16 DWORDs (one row) insert 4 DWORDs -> LDS row stride 40 h
// ---------------------------------------------------------------------------
__device__ __forceinline__ void tdm_load_b_tile(const unsigned short* wbf,
                                                int kc, unsigned int ldsOff) {
  const unsigned long long ga =
      (unsigned long long)(uintptr_t)wbf + (unsigned long long)kc * 2ull;

  u32x4 g0;
  g0.x = 1u;                                             // count=1 (valid D#)
  g0.y = ldsOff;                                         // lds_addr (bytes)
  g0.z = (unsigned int)(ga & 0xFFFFFFFFull);             // global_addr[31:0]
  g0.w = (unsigned int)((ga >> 32) & 0x01FFFFFFull)      // global_addr[56:32]
       | (2u << 30);                                     // type = 2 ("image")

  i32x8 g1;
  g1.s0 = (int)((1u << 16)      // data_size = 2 bytes (bf16)
              | (1u << 20)      // pad_enable
              | (3u << 22)      // pad_interval code 3 = 16 DWORDs (64B row)
              | (3u << 25));    // pad_amount  code 3 = 4 DWORDs  (8 halves)
  g1.s1 = (int)(((unsigned)K_TOTAL & 0xFFFFu) << 16);    // tensor_dim0 lo
  g1.s2 = (int)((((unsigned)K_TOTAL >> 16) & 0xFFFFu)    // tensor_dim0 hi
              | (((unsigned)OUT_DIM & 0xFFFFu) << 16));  // tensor_dim1 lo
  g1.s3 = (int)(((unsigned)OUT_DIM >> 16)                // tensor_dim1 hi
              | ((unsigned)K_CHUNK << 16));              // tile_dim0 = 32
  g1.s4 = (int)((unsigned)OUT_DIM & 0xFFFFu);            // tile_dim1=512, d2=0
  g1.s5 = (int)(unsigned)K_TOTAL;                        // tensor_dim0_stride
  g1.s6 = 0;
  g1.s7 = 0;

  i32x4 gz4 = {0, 0, 0, 0};                              // groups 2/3 unused
  i32x8 gz8 = {0, 0, 0, 0, 0, 0, 0, 0};                  // (2D tensor, NULL)
  __builtin_amdgcn_tensor_load_to_lds(g0, g1, gz4, gz4, gz8, 0);
}

// ---------------------------------------------------------------------------
// Main GEMM kernel (TDM + double-buffered pipeline, 1 barrier per chunk)
// LDS layout (halves): [B0 | B1 | A0 | A1]  = 102,400 B (< 320KB WGP)
// ---------------------------------------------------------------------------
__global__ __launch_bounds__(THREADS, 1)
void kan_cheby_wmma_tdm(const float* __restrict__ x,
                        const unsigned short* __restrict__ wbf,
                        float* __restrict__ out) {
  extern __shared__ unsigned short smem[];

  const int tid    = threadIdx.x;
  const int lane   = tid & 31;
  const int wave   = tid >> 5;
  const int mwave  = wave & 1;
  const int nwave  = wave >> 1;
  const int r0     = blockIdx.x * M_BLK;

  const int lane15  = lane & 15;
  const int halfSel = lane >> 4;
  const int koff    = halfSel * 8;

  const int aRowF = tid >> 2;          // 128 rows, 4 threads/row
  const int aColF = (tid & 3) * 8;     // 8 cols/thread

  v8f acc[4][4];
#pragma unroll
  for (int mt = 0; mt < 4; ++mt)
#pragma unroll
    for (int nt = 0; nt < 4; ++nt)
      acc[mt][nt] = v8f{0.f, 0.f, 0.f, 0.f, 0.f, 0.f, 0.f, 0.f};

  // ---- prologue: stage chunk 0 ----
  if (wave == 0)
    tdm_load_b_tile(wbf, 0, (unsigned int)(uintptr_t)(void*)&smem[0]);
  store_a_tile(&smem[2 * BSZ + aRowF * LDS_LD + aColF],
               load_a_src(0, x, r0, aRowF, aColF));
  if (wave == 0)
    __builtin_amdgcn_s_wait_tensorcnt(0);
  __syncthreads();

  for (int it = 0; it < NCHUNK; ++it) {
    const unsigned int bCur = (unsigned)(it & 1) * BSZ;
    const unsigned int bNxt = (unsigned)((it + 1) & 1) * BSZ;
    const unsigned int aCur = 2u * BSZ + (unsigned)(it & 1) * ASZ;
    const unsigned int aNxt = 2u * BSZ + (unsigned)((it + 1) & 1) * ASZ;

    // ---- stage chunk it+1: DMA B now, load x now, convert later ----
    ASrc nextA;
    const bool havePrefetch = (it + 1 < NCHUNK);
    if (havePrefetch) {
      if (wave == 0)
        tdm_load_b_tile(wbf, (it + 1) * K_CHUNK,
                        (unsigned int)(uintptr_t)(void*)&smem[bNxt]);
      nextA = load_a_src((it + 1) * K_CHUNK, x, r0, aRowF, aColF);
    }

    // ---- WMMA: 64x64 per wave, K = 32 (pure DScnt waits) ----
    v16bf afrag[4];
#pragma unroll
    for (int mt = 0; mt < 4; ++mt) {
      const int aRow = mwave * 64 + mt * 16 + lane15;
      afrag[mt] = load_frag16(&smem[aCur + aRow * LDS_LD + koff]);
    }
#pragma unroll
    for (int nt = 0; nt < 4; ++nt) {
      const int bRow = nwave * 64 + nt * 16 + lane15;
      v16bf bfrag = load_frag16(&smem[bCur + bRow * LDS_LD + koff]);
#pragma unroll
      for (int mt = 0; mt < 4; ++mt) {
        acc[mt][nt] = __builtin_amdgcn_wmma_f32_16x16x32_bf16(
            false, afrag[mt], false, bfrag, (short)0, acc[mt][nt],
            false, false);
      }
    }

    // ---- finish staging chunk it+1 (conversion + ds_store) ----
    if (havePrefetch)
      store_a_tile(&smem[aNxt + aRowF * LDS_LD + aColF], nextA);

    if (wave == 0)
      __builtin_amdgcn_s_wait_tensorcnt(0);   // next B tile landed
    __syncthreads();                           // next A visible, bufs reusable
  }

  // ---- writeback (f32 C layout: VGPR e -> M=e / M=e+8) ----
#pragma unroll
  for (int mt = 0; mt < 4; ++mt) {
#pragma unroll
    for (int nt = 0; nt < 4; ++nt) {
      const int colg = nwave * 64 + nt * 16 + lane15;
#pragma unroll
      for (int e = 0; e < 8; ++e) {
        const int rowg = r0 + mwave * 64 + mt * 16 + halfSel * 8 + e;
        out[(size_t)rowg * OUT_DIM + colg] = acc[mt][nt][e];
      }
    }
  }
}

// ---------------------------------------------------------------------------
// Fallback (no workspace): fused kernel, weights converted in-loop
// ---------------------------------------------------------------------------
__global__ __launch_bounds__(THREADS, 1)
void kan_cheby_wmma_fused(const float* __restrict__ x,
                          const float* __restrict__ wb,
                          const float* __restrict__ wc,
                          float* __restrict__ out) {
  __shared__ unsigned short As[M_BLK * LDS_LD];
  __shared__ unsigned short Bs[N_BLK * LDS_LD];

  const int tid    = threadIdx.x;
  const int lane   = tid & 31;
  const int wave   = tid >> 5;
  const int mwave  = wave & 1;
  const int nwave  = wave >> 1;
  const int r0     = blockIdx.x * M_BLK;
  const int lane15  = lane & 15;
  const int halfSel = lane >> 4;
  const int koff    = halfSel * 8;
  const int aRowF   = tid >> 2;
  const int aColF   = (tid & 3) * 8;

  v8f acc[4][4];
#pragma unroll
  for (int mt = 0; mt < 4; ++mt)
#pragma unroll
    for (int nt = 0; nt < 4; ++nt)
      acc[mt][nt] = v8f{0.f, 0.f, 0.f, 0.f, 0.f, 0.f, 0.f, 0.f};

  for (int kc = 0; kc < K_TOTAL; kc += K_CHUNK) {
    __syncthreads();
    store_a_tile(&As[aRowF * LDS_LD + aColF],
                 load_a_src(kc, x, r0, aRowF, aColF));
    {
      const int n = tid;
      const float* src;
      float scale;
      if (kc < IN_DIM) {
        src = wb + (size_t)n * IN_DIM + kc;
        scale = 1.0f;
      } else {
        src = wc + (size_t)n * (IN_DIM * ORDER) + (kc - IN_DIM);
        scale = 0.1f;
      }
#pragma unroll
      for (int q = 0; q < 8; ++q) {
        f32x4 v = *(const f32x4*)(src + q * 4);
#pragma unroll
        for (int j = 0; j < 4; ++j)
          Bs[n * LDS_LD + q * 4 + j] = f2bf(v[j] * scale);
      }
    }
    __syncthreads();

    v16bf afrag[4];
#pragma unroll
    for (int mt = 0; mt < 4; ++mt) {
      const int aRow = mwave * 64 + mt * 16 + lane15;
      afrag[mt] = load_frag16(&As[aRow * LDS_LD + koff]);
    }
#pragma unroll
    for (int nt = 0; nt < 4; ++nt) {
      const int bRow = nwave * 64 + nt * 16 + lane15;
      v16bf bfrag = load_frag16(&Bs[bRow * LDS_LD + koff]);
#pragma unroll
      for (int mt = 0; mt < 4; ++mt) {
        acc[mt][nt] = __builtin_amdgcn_wmma_f32_16x16x32_bf16(
            false, afrag[mt], false, bfrag, (short)0, acc[mt][nt],
            false, false);
      }
    }
  }

#pragma unroll
  for (int mt = 0; mt < 4; ++mt) {
#pragma unroll
    for (int nt = 0; nt < 4; ++nt) {
      const int colg = nwave * 64 + nt * 16 + lane15;
#pragma unroll
      for (int e = 0; e < 8; ++e) {
        const int rowg = r0 + mwave * 64 + mt * 16 + halfSel * 8 + e;
        out[(size_t)rowg * OUT_DIM + colg] = acc[mt][nt][e];
      }
    }
  }
}

extern "C" void kernel_launch(void* const* d_in, const int* in_sizes, int n_in,
                              void* d_out, int out_size, void* d_ws, size_t ws_size,
                              hipStream_t stream) {
  (void)in_sizes; (void)n_in; (void)out_size;
  const float* x  = (const float*)d_in[0];   // (16384, 512)
  const float* wb = (const float*)d_in[1];   // (512, 512)
  const float* wc = (const float*)d_in[2];   // (512, 512, 10) viewed (512, 5120)
  float* out = (float*)d_out;                // (16384, 512)

  if (d_ws != nullptr && ws_size >= WS_BYTES) {
    unsigned short* wbf = (unsigned short*)d_ws;
    {
      const int total = OUT_DIM * K_TOTAL / 4;           // 720896 quads
      dim3 grid((total + 255) / 256), block(256);
      convert_weights<<<grid, block, 0, stream>>>(wb, wc, wbf);
    }
    {
      dim3 grid(B_ROWS / M_BLK);                         // 128 workgroups
      dim3 block(THREADS);                               // 16 wave32
      const size_t lds = (size_t)(2 * BSZ + 2 * ASZ) *
                         sizeof(unsigned short);         // 102,400 B
      kan_cheby_wmma_tdm<<<grid, block, lds, stream>>>(x, wbf, out);
    }
  } else {
    dim3 grid(B_ROWS / M_BLK);
    dim3 block(THREADS);
    kan_cheby_wmma_fused<<<grid, block, 0, stream>>>(x, wb, wc, out);
  }
}
